// ResidualVectorQuantizer_78683800862861
// MI455X (gfx1250) — matrix operand order repaired and checked by reading, not compiled, not run on hardware
//
#include <hip/hip_runtime.h>
#include <hip/hip_bf16.h>

// ---------------- problem constants (match reference) ----------------
static constexpr int QN = 8;       // NUM_Q
static constexpr int KC = 1024;    // CODEBOOK_SIZE
static constexpr int D  = 512;     // HIDDEN
static constexpr int B  = 16;
static constexpr int T  = 1024;
static constexpr int N  = B * T;   // 16384 rows

typedef __attribute__((ext_vector_type(16))) _Float16 v16h;
typedef __attribute__((ext_vector_type(8)))  _Float16 v8h;
typedef __attribute__((ext_vector_type(4)))  _Float16 v4h;
typedef __attribute__((ext_vector_type(8)))  float    v8f;
typedef __attribute__((ext_vector_type(4)))  float    v4f;

// ---- CDNA5 async global->LDS copy (ASYNCcnt-tracked) ----------------
// VGLOBAL encoding: vdst = per-lane LDS byte address, vaddr = 64-bit
// global address, saddr = off. Generic->LDS address = low 32 bits of
// the flat address (ISA 10.2).
__device__ __forceinline__ void async_copy_b128(const void* g, void* l) {
  unsigned ldso = (unsigned)(size_t)l;
  asm volatile("global_load_async_to_lds_b128 %0, %1, off"
               :: "v"(ldso), "v"(g) : "memory");
}

#if __has_builtin(__builtin_amdgcn_s_wait_asynccnt)
#define WAIT_ASYNC0() __builtin_amdgcn_s_wait_asynccnt(0)
#else
#define WAIT_ASYNC0() asm volatile("s_wait_asynccnt 0" ::: "memory")
#endif

// ---------------------------------------------------------------------
// Kernel 0: convert full codebook f32 -> f16 once (row-major preserved).
// ---------------------------------------------------------------------
__global__ __launch_bounds__(256) void rvq_cvt_cb(const float* __restrict__ cb,
                                                  _Float16* __restrict__ cbh) {
  const size_t i = (size_t)(blockIdx.x * blockDim.x + threadIdx.x) * 4;
  v4f v = *(const v4f*)(cb + i);
  v4h h;
  h[0] = (_Float16)v[0]; h[1] = (_Float16)v[1];
  h[2] = (_Float16)v[2]; h[3] = (_Float16)v[3];
  *(v4h*)(cbh + i) = h;
}

// ---------------------------------------------------------------------
// Kernel 1: codebook squared norms, one wave32 per codebook row.
// ---------------------------------------------------------------------
__global__ __launch_bounds__(256) void rvq_cbnorm(const float* __restrict__ cb,
                                                  float* __restrict__ cbn) {
  const int gwave = (blockIdx.x * blockDim.x + threadIdx.x) >> 5;  // row id
  const int lane  = threadIdx.x & 31;
  const float* row = cb + (size_t)gwave * D;
  float s = 0.0f;
  for (int d = lane * 4; d < D; d += 32 * 4) {
    v4f v = *(const v4f*)(row + d);
    s += v[0] * v[0] + v[1] * v[1] + v[2] * v[2] + v[3] * v[3];
  }
  #pragma unroll
  for (int m = 16; m >= 1; m >>= 1) s += __shfl_xor(s, m, 32);
  if (lane == 0) cbn[gwave] = s;
}

// ---------------------------------------------------------------------
// Kernel 2: per-stage distance + argmin via WMMA f16 (f32 accumulate).
//  - residual tile (z - quant) converted once to f16 in LDS, then the 16
//    A fragments (16x32 each, documented CDNA5 lane layout) are hoisted
//    into registers for the entire code sweep.
//  - B (f16 codebook) tiles are double-buffered in LDS via async
//    global->LDS DMA, shared by all 4 waves of the block.
//  - argmin over score = ||c_k||^2 - 2*r.c_k  (rnorm constant per row).
// ---------------------------------------------------------------------
__global__ __launch_bounds__(128) void rvq_dist_argmin(
    const float* __restrict__ z, const float* __restrict__ quant,
    const _Float16* __restrict__ cbh, const float* __restrict__ cbn,
    int* __restrict__ idxOut, float* __restrict__ idxFOut, int q) {
  __shared__ __attribute__((aligned(32))) _Float16 smA[4 * 16 * D];  // 64 KB
  __shared__ __attribute__((aligned(32))) _Float16 smB[2 * 16 * D];  // 32 KB

  const int tid  = threadIdx.x;
  const int wave = tid >> 5;
  const int lane = tid & 31;
  const int grp  = lane >> 4;   // 0: lanes 0-15, 1: lanes 16-31
  const int ln   = lane & 15;
  const int m0   = (blockIdx.x * 4 + wave) * 16;
  _Float16* myA  = &smA[wave * 16 * D];

  // ---- Phase 1: residual tile -> f16 in LDS (lane: row=ln, half of D) ----
  {
    const int d0 = grp * (D / 2);
    const float* zp = z     + (size_t)(m0 + ln) * D + d0;
    const float* qp = quant + (size_t)(m0 + ln) * D + d0;
    _Float16* op = myA + ln * D + d0;
    for (int d = 0; d < D / 2; d += 4) {
      v4f zv = *(const v4f*)(zp + d);
      v4f qv = *(const v4f*)(qp + d);
      op[d + 0] = (_Float16)(zv[0] - qv[0]);
      op[d + 1] = (_Float16)(zv[1] - qv[1]);
      op[d + 2] = (_Float16)(zv[2] - qv[2]);
      op[d + 3] = (_Float16)(zv[3] - qv[3]);
    }
  }
  __syncthreads();

  // ---- Hoist the 16 A fragments into registers (128 VGPRs, wave32) ----
  // CDNA5 16-bit A layout: lane row M=ln; halves 0-7 at d=kk+8*grp,
  // halves 8-15 at d=kk+8*grp+16.
  v16h a[16];
  #pragma unroll
  for (int c = 0; c < 16; ++c) {
    const _Float16* arow = myA + ln * D + c * 32 + grp * 8;
    v8h lo = *(const v8h*)(arow);
    v8h hi = *(const v8h*)(arow + 16);
    #pragma unroll
    for (int i = 0; i < 8; ++i) { a[c][i] = lo[i]; a[c][8 + i] = hi[i]; }
  }

  // ---- Prime B double-buffer: async DMA tile 0 (16 codes x 512 f16) ----
  {
    const char* g = (const char*)cbh;
    char* l = (char*)smB;
    #pragma unroll
    for (int r = 0; r < 8; ++r) {
      const int off = (tid + r * 128) * 16;   // 16 KB / 128 threads
      async_copy_b128(g + off, l + off);
    }
  }
  WAIT_ASYNC0();
  __syncthreads();

  float bestD[8];
  int   bestK[8];
  #pragma unroll
  for (int j = 0; j < 8; ++j) { bestD[j] = 3.4e38f; bestK[j] = 0; }

  for (int kt = 0; kt < KC / 16; ++kt) {
    // Issue async DMA for the next tile into the alternate buffer.
    if (kt + 1 < KC / 16) {
      const char* g = (const char*)(cbh + (size_t)(kt + 1) * 16 * D);
      char* l = (char*)(smB + ((kt + 1) & 1) * 16 * D);
      #pragma unroll
      for (int r = 0; r < 8; ++r) {
        const int off = (tid + r * 128) * 16;
        async_copy_b128(g + off, l + off);
      }
    }

    // Compute on the current tile: 16 chained WMMAs over D=512.
    const _Float16* bt = smB + (kt & 1) * 16 * D;
    v8f acc = {};
    #pragma unroll
    for (int c = 0; c < 16; ++c) {
      // B layout: lane column N=ln; halves 0-7 at d=c*32+8*grp, 8-15 at +16.
      const _Float16* brow = bt + ln * D + c * 32 + grp * 8;
      v8h blo = *(const v8h*)(brow);
      v8h bhi = *(const v8h*)(brow + 16);
      v16h b;
      #pragma unroll
      for (int i = 0; i < 8; ++i) { b[i] = blo[i]; b[8 + i] = bhi[i]; }
      acc = __builtin_amdgcn_wmma_f32_16x16x32_f16(
          false, a[c], false, b, (short)0, acc, false, false);
    }

    const int code = kt * 16 + ln;
    const float cn = cbn[code];
    #pragma unroll
    for (int j = 0; j < 8; ++j) {
      const float d2 = cn - 2.0f * acc[j];
      if (d2 < bestD[j]) { bestD[j] = d2; bestK[j] = code; }
    }

    // Next tile's DMA must land before anyone reads it.
    WAIT_ASYNC0();
    __syncthreads();
  }

  // ---- Reduce argmin across the 16 column-lanes (same grp: masks 1..8) ----
  #pragma unroll
  for (int j = 0; j < 8; ++j) {
    float v = bestD[j];
    int   k = bestK[j];
    #pragma unroll
    for (int m = 8; m >= 1; m >>= 1) {
      const float ov = __shfl_xor(v, m, 32);
      const int   ok = __shfl_xor(k, m, 32);
      if (ov < v || (ov == v && ok < k)) { v = ov; k = ok; }
    }
    if (ln == 0) {
      const int m = m0 + j + 8 * grp;
      idxOut[m] = k;
      const int bb = m / T, tt = m % T;
      idxFOut[((size_t)bb * QN + q) * T + tt] = (float)k;  // indices [B,Q,T]
    }
  }
}

// ---------------------------------------------------------------------
// Kernel 3: gather chosen code (fp32), quant += code, loss accumulation.
// ---------------------------------------------------------------------
__global__ __launch_bounds__(256) void rvq_update(
    const float* __restrict__ z, float* __restrict__ quant,
    const float* __restrict__ cb, const int* __restrict__ idx,
    float* __restrict__ lossAcc) {
  const int wave = threadIdx.x >> 5;
  const int lane = threadIdx.x & 31;
  const int n = blockIdx.x * 8 + wave;
  const int k = idx[n];
  const float* crow = cb + (size_t)k * D;
  const float* zrow = z + (size_t)n * D;
  float* qrow = quant + (size_t)n * D;
  float part = 0.0f;
  for (int d = lane * 4; d < D; d += 32 * 4) {
    v4f c  = *(const v4f*)(crow + d);
    v4f zv = *(const v4f*)(zrow + d);
    v4f qv = *(const v4f*)(qrow + d);
    #pragma unroll
    for (int i = 0; i < 4; ++i) {
      const float r = zv[i] - qv[i];      // residual before this stage
      const float df = c[i] - r;
      part += df * df;
      qv[i] = qv[i] + c[i];
    }
    *(v4f*)(qrow + d) = qv;
  }
  #pragma unroll
  for (int m = 16; m >= 1; m >>= 1) part += __shfl_xor(part, m, 32);
  if (lane == 0) atomicAdd(lossAcc, part);
}

__global__ void rvq_finish(const float* __restrict__ lossAcc,
                           float* __restrict__ lossOut) {
  lossOut[0] = lossAcc[0] * (1.0f / ((float)N * (float)D * (float)QN));
}

// ---------------------------------------------------------------------
extern "C" void kernel_launch(void* const* d_in, const int* in_sizes, int n_in,
                              void* d_out, int out_size, void* d_ws, size_t ws_size,
                              hipStream_t stream) {
  const float* z   = (const float*)d_in[0];         // [B,T,D]
  const float* cbs = (const float*)d_in[1];         // [Q,K,D]

  float* out   = (float*)d_out;
  float* quant = out;                               // [B,T,D] = N*D floats
  float* idxF  = out + (size_t)N * D;               // [B,Q,T]
  float* lossO = idxF + (size_t)B * QN * T;         // scalar

  _Float16* cbh  = (_Float16*)d_ws;                 // QN*KC*D f16 (8 MB)
  float* cbn     = (float*)(cbh + (size_t)QN * KC * D);  // QN*KC floats
  int*   idx     = (int*)(cbn + (size_t)QN * KC);   // N ints
  float* lossAcc = (float*)(idx + N);               // 1 float

  hipMemsetAsync(quant, 0, (size_t)N * D * sizeof(float), stream);
  hipMemsetAsync(lossAcc, 0, sizeof(float), stream);

  // One-time: f16 codebook + all Q*K codebook norms.
  rvq_cvt_cb<<<(QN * KC * D / 4) / 256, 256, 0, stream>>>(cbs, cbh);
  rvq_cbnorm<<<(QN * KC) / 8, 256, 0, stream>>>(cbs, cbn);

  for (int q = 0; q < QN; ++q) {
    rvq_dist_argmin<<<N / 64, 128, 0, stream>>>(
        z, quant, cbh + (size_t)q * KC * D, cbn + (size_t)q * KC, idx, idxF, q);
    rvq_update<<<N / 8, 256, 0, stream>>>(z, quant, cbs + (size_t)q * KC * D,
                                          idx, lossAcc);
  }
  rvq_finish<<<1, 1, 0, stream>>>(lossAcc, lossO);
}
